// SuperONN2d_v2_12360915878419
// MI455X (gfx1250) — matrix operand order repaired and checked by reading, not compiled, not run on hardware
//
#include <hip/hip_runtime.h>
#include <hip/hip_bf16.h>

typedef __attribute__((ext_vector_type(16))) _Float16 v16h;
typedef __attribute__((ext_vector_type(8)))  float    v8f;
typedef __attribute__((ext_vector_type(4)))  int      v4i;

#define CIN   32
#define COUT  64
#define Hh    128
#define Ww    128
#define HW    (Hh*Ww)
#define Bn    4
#define KK    9
#define CK    864      // 96 channels * 9 taps = GEMM K
#define NT    32       // pixels per workgroup (one row segment)
#define KC    32       // K per WMMA chunk
#define NCH   (CK/KC)  // 27 chunks
#define MT    (COUT/16) // 4 cout tiles

// LDS layout (dynamic): [ Bpanel 55296 B ][ corners 9216 B | Abuf 16384 B (overlap) ]
#define BPANEL_BYTES (NT*CK*2)          // 55296
#define CIDX_BYTES   (NT*KK*4*4)        // 4608
#define SMEM_BYTES   (BPANEL_BYTES + 16384)   // 71680

#if defined(__gfx1250__) && __has_builtin(__builtin_amdgcn_global_load_async_to_lds_b128) && __has_builtin(__builtin_amdgcn_s_wait_asynccnt)
#define USE_ASYNC_LDS 1
#else
#define USE_ASYNC_LDS 0
#endif

#define AS1 __attribute__((address_space(1)))
#define AS3 __attribute__((address_space(3)))

__device__ __forceinline__ void async_copy32(const void* g, void* l) {
#if USE_ASYNC_LDS
  __builtin_amdgcn_global_load_async_to_lds_b128((AS1 v4i*)g, (AS3 v4i*)l, 0, 0);
  __builtin_amdgcn_global_load_async_to_lds_b128((AS1 v4i*)((const char*)g + 16),
                                                 (AS3 v4i*)((char*)l + 16), 0, 0);
#else
  (void)g; (void)l;
#endif
}

// ---------------- Kernel 0: pack weights fp32 -> f16 in WMMA A-fragment order ----------------
// pA[((mt*NCH+cc)*32+lane)*16 + j] = (f16) W[mt*16 + (lane&15)][cc*32 + Koff(lane,j)]
// Koff = (lane>>4)*16 + j   (contiguous K order, matches B fragment)
__global__ void pack_weights_kernel(const float* __restrict__ w, _Float16* __restrict__ pA) {
  int idx = blockIdx.x * blockDim.x + threadIdx.x;
  if (idx >= MT * NCH * 32 * 16) return;
  int j    = idx & 15;
  int lane = (idx >> 4) & 31;
  int cc   = (idx >> 9) % NCH;
  int mt   = idx / (NCH * 512);
  int m = lane & 15, hi = lane >> 4;
  int K = cc * KC + hi * 16 + j;
  pA[idx] = (_Float16)w[(size_t)(mt * 16 + m) * CK + K];
}

// ---------------- Kernel 1: shift conv (32 -> 2, 3x3, pad 1) + clip ----------------
__global__ void shift_conv_kernel(const float* __restrict__ x,
                                  const float* __restrict__ sw,
                                  const float* __restrict__ sb,
                                  float* __restrict__ shifts) {
  int idx = blockIdx.x * blockDim.x + threadIdx.x;
  if (idx >= Bn * HW) return;
  int b   = idx >> 14;
  int pix = idx & (HW - 1);
  int oy = pix >> 7, ox = pix & (Ww - 1);
  float a0 = sb[0], a1 = sb[1];
  const float* xb = x + (size_t)b * CIN * HW;
  for (int ci = 0; ci < CIN; ++ci) {
    const float* xc = xb + ci * HW;
    #pragma unroll
    for (int ky = 0; ky < 3; ++ky) {
      int yy = oy - 1 + ky;
      if (yy < 0 || yy >= Hh) continue;
      #pragma unroll
      for (int kx = 0; kx < 3; ++kx) {
        int xx = ox - 1 + kx;
        if (xx < 0 || xx >= Ww) continue;
        float xv = xc[yy * Ww + xx];
        int wi = ci * 9 + ky * 3 + kx;
        a0 = fmaf(xv, sw[wi], a0);
        a1 = fmaf(xv, sw[CIN * 9 + wi], a1);
      }
    }
  }
  a0 = fminf(fmaxf(a0, -4.f), 4.f);
  a1 = fminf(fmaxf(a1, -4.f), 4.f);
  shifts[((size_t)b * 2 + 0) * HW + pix] = a0;
  shifts[((size_t)b * 2 + 1) * HW + pix] = a1;
}

// ---------------- Kernel 2: fused deform-im2col + WMMA GEMM ----------------
__global__ void __launch_bounds__(256)
deform_wmma_kernel(const float* __restrict__ x,
                   const _Float16* __restrict__ packedA,
                   const float* __restrict__ bias,
                   const float* __restrict__ shifts,
                   float* __restrict__ out) {
  extern __shared__ char smem[];
  _Float16* Bpanel = (_Float16*)smem;                          // [pixel][K], 55296 B
  int*      cidx   = (int*)(smem + BPANEL_BYTES);              // 4608 B  (phases 1-2)
  float*    cwt    = (float*)(smem + BPANEL_BYTES + CIDX_BYTES); // 4608 B (phases 1-2)
  _Float16* Abuf   = (_Float16*)(smem + BPANEL_BYTES);         // 16384 B (phase 3, overlaps)

  const int b       = blockIdx.x >> 9;        // HW/NT = 512 tiles per batch
  const int tileInB = blockIdx.x & 511;
  const int pixbase = tileInB * NT;           // whole tile lies in one row
  const int oy      = pixbase >> 7;
  const int oxbase  = pixbase & (Ww - 1);

  // ---- Phase 1: per (pixel, tap) bilinear corner indices & weights ----
  for (int t = threadIdx.x; t < NT * KK; t += blockDim.x) {
    int p = t / KK, k = t % KK;
    int ox = oxbase + p;
    float s0 = shifts[((size_t)b * 2 + 0) * HW + oy * Ww + ox];
    float s1 = shifts[((size_t)b * 2 + 1) * HW + oy * Ww + ox];
    int ky = k / 3, kx = k % 3;
    // offset = repeat(shifts, 9, ch): offset ch m -> shifts[m/9]; dy=ch 2k, dx=ch 2k+1
    float dy = ((2 * k)     / 9) ? s1 : s0;
    float dx = ((2 * k + 1) / 9) ? s1 : s0;
    float py = (float)(oy - 1 + ky) + dy;
    float px = (float)(ox - 1 + kx) + dx;
    float y0f = floorf(py), x0f = floorf(px);
    float ly = py - y0f, lx = px - x0f;
    int y0 = (int)y0f, x0 = (int)x0f;
    #pragma unroll
    for (int cy = 0; cy < 2; ++cy)
      #pragma unroll
      for (int cx = 0; cx < 2; ++cx) {
        int yc = y0 + cy, xc = x0 + cx;
        bool valid = (yc >= 0) && (yc < Hh) && (xc >= 0) && (xc < Ww);
        int yi = min(max(yc, 0), Hh - 1);
        int xi = min(max(xc, 0), Ww - 1);
        float w = (cy ? ly : 1.f - ly) * (cx ? lx : 1.f - lx);
        int slot = t * 4 + cy * 2 + cx;
        cidx[slot] = yi * Ww + xi;
        cwt[slot]  = valid ? w : 0.f;
      }
  }
  __syncthreads();

  // ---- Phase 2: build full 864x32 B panel in f16 ----
  // One 4-corner gather of x per (cin, tap, pixel) yields all 3 power rows.
  const float* xb = x + (size_t)b * CIN * HW;
  for (int it = 0; it < (NT * KK * CIN) / 256; ++it) {   // 36 iterations
    int t    = it * 256 + threadIdx.x;
    int p    = t & (NT - 1);
    int rest = t >> 5;
    int k    = rest % KK;
    int cin  = rest / KK;
    int slot = (p * KK + k) * 4;
    const float* xc = xb + cin * HW;
    float v1 = 0.f, v2 = 0.f, v3 = 0.f;
    #pragma unroll
    for (int j = 0; j < 4; ++j) {
      float w  = cwt[slot + j];
      float xv = xc[cidx[slot + j]];
      v1 = fmaf(w, xv, v1);
      float x2 = xv * xv;
      v2 = fmaf(w, x2, v2);
      v3 = fmaf(w, x2 * xv, v3);
    }
    int rb = cin * KK + k;                 // K row = (q*32+cin)*9+k = q*288 + rb
    _Float16* bp = &Bpanel[p * CK];
    bp[rb      ] = (_Float16)v1;
    bp[rb + 288] = (_Float16)v2;
    bp[rb + 576] = (_Float16)v3;
  }
  __syncthreads();   // after this, corner tables dead -> Abuf may reuse their LDS

  // ---- Phase 3: 8 waves = 4 cout-tiles x 2 pixel-tiles, K-loop WMMA ----
  const int wave = threadIdx.x >> 5;
  const int lane = threadIdx.x & 31;
  const int mt   = wave >> 1;          // cout tile 0..3
  const int nt   = wave & 1;           // pixel tile 0..1
  const int m    = lane & 15;
  const int hi   = lane >> 4;
  const int pcol = nt * 16 + m;        // B-fragment column (pixel)
  // packed A bytes: fragment (mt,cc,lane) at ((mt*NCH+cc)*32+lane)*32
  const char* pAbase = (const char*)packedA + (size_t)mt * NCH * 1024;

#if USE_ASYNC_LDS
  char* buf0 = (char*)Abuf + (size_t)wave * 1024;        // per-wave double buffer
  char* buf1 = (char*)Abuf + (size_t)(8 + wave) * 1024;
  async_copy32(pAbase + lane * 32, buf0 + lane * 32);    // preload chunk 0
#endif

  v8f acc = {};
  for (int cc = 0; cc < NCH; ++cc) {
#if USE_ASYNC_LDS
    char* cur = (cc & 1) ? buf1 : buf0;
    if (cc + 1 < NCH) {
      char* nxt = (cc & 1) ? buf0 : buf1;
      async_copy32(pAbase + (size_t)(cc + 1) * 1024 + lane * 32, nxt + lane * 32);
      __builtin_amdgcn_s_wait_asynccnt(2);   // current buffer's 2 loads complete
    } else {
      __builtin_amdgcn_s_wait_asynccnt(0);
    }
    v16h a = *(const v16h*)(cur + lane * 32);
#else
    v16h a = *(const v16h*)(pAbase + (size_t)cc * 1024 + lane * 32);
    if (cc + 1 < NCH)
      __builtin_prefetch(pAbase + (size_t)(cc + 1) * 1024 + lane * 32, 0, 0);
#endif
    // B fragment (32x16 f16): lanes 0-15 hold K 0-15, lanes 16-31 hold K 16-31
    v16h bf = *(const v16h*)&Bpanel[pcol * CK + cc * KC + hi * 16];
    acc = __builtin_amdgcn_wmma_f32_16x16x32_f16(false, a, false, bf,
                                                 (short)0, acc, false, false);
  }

  // D layout: VGPR r, lanes 0-15: M=r ; lanes 16-31: M=r+8 ; N=lane&15
  #pragma unroll
  for (int r = 0; r < 8; ++r) {
    int co  = mt * 16 + r + 8 * hi;
    int n   = lane & 15;
    int px_ = pixbase + nt * 16 + n;
    out[((size_t)b * COUT + co) * HW + px_] = acc[r] + bias[co];
  }
}

extern "C" void kernel_launch(void* const* d_in, const int* in_sizes, int n_in,
                              void* d_out, int out_size, void* d_ws, size_t ws_size,
                              hipStream_t stream) {
  const float* x        = (const float*)d_in[0];
  const float* weight   = (const float*)d_in[1];
  const float* bias     = (const float*)d_in[2];
  const float* shift_w  = (const float*)d_in[3];
  const float* shift_b  = (const float*)d_in[4];
  float* out = (float*)d_out;

  // workspace: [ shifts: B*2*H*W f32 = 524288 B ][ packedA: 55296 f16 = 110592 B ]
  float*    shifts  = (float*)d_ws;
  _Float16* packedA = (_Float16*)((char*)d_ws + (size_t)Bn * 2 * HW * 4);

  (void)in_sizes; (void)n_in; (void)out_size; (void)ws_size;

  pack_weights_kernel<<<(MT * NCH * 512 + 255) / 256, 256, 0, stream>>>(weight, packedA);
  shift_conv_kernel<<<(Bn * HW) / 256, 256, 0, stream>>>(x, shift_w, shift_b, shifts);
  deform_wmma_kernel<<<Bn * (HW / NT), 256, SMEM_BYTES, stream>>>(x, packedA, bias, shifts, out);
}
